// Fp8Linear_65738769432996
// MI455X (gfx1250) — 8-bit matrix operand order repaired and checked by reading, compiled, not run on hardware
//
#include <hip/hip_runtime.h>

constexpr int Mdim = 2048;
constexpr int Kdim = 4096;
constexpr int Ndim = 4096;
constexpr float FP8_MAX = 448.0f;

typedef int   v16i __attribute__((ext_vector_type(16)));
typedef float v8f  __attribute__((ext_vector_type(8)));

__global__ void init_amax_kernel(unsigned int* amax) {
    if (threadIdx.x < 2) amax[threadIdx.x] = 0u;
}

__global__ __launch_bounds__(256) void amax_kernel(const float* __restrict__ in,
                                                   size_t n4,
                                                   unsigned int* __restrict__ slot) {
    __shared__ float red[256];
    const float4* in4 = (const float4*)in;
    float m = 0.0f;
    for (size_t i = (size_t)blockIdx.x * blockDim.x + threadIdx.x; i < n4;
         i += (size_t)gridDim.x * blockDim.x) {
        float4 v = in4[i];
        m = fmaxf(m, fabsf(v.x));
        m = fmaxf(m, fabsf(v.y));
        m = fmaxf(m, fabsf(v.z));
        m = fmaxf(m, fabsf(v.w));
    }
    red[threadIdx.x] = m;
    __syncthreads();
    for (int s = 128; s > 0; s >>= 1) {
        if ((int)threadIdx.x < s)
            red[threadIdx.x] = fmaxf(red[threadIdx.x], red[threadIdx.x + s]);
        __syncthreads();
    }
    if (threadIdx.x == 0) atomicMax(slot, __float_as_uint(red[0]));
}

__device__ __forceinline__ unsigned int fp8_e4m3_bits(float f) {
    unsigned int u    = __float_as_uint(f);
    unsigned int sign = (u >> 31) << 7;
    unsigned int absu = u & 0x7fffffffu;
    unsigned int bits;
    if (absu >= 0x3C800000u) {
        unsigned int lsb     = (absu >> 20) & 1u;
        unsigned int rounded = absu + 0x0007FFFFu + lsb;
        int e            = (int)(rounded >> 23) - 127 + 7;
        unsigned int man = (rounded >> 20) & 7u;
        if (e > 15) { e = 15; man = 6u; }
        bits = ((unsigned int)e << 3) | man;
    } else {
        float q = rintf(__uint_as_float(absu) * 512.0f);
        unsigned int man = (unsigned int)q;
        bits = man;
    }
    return sign | bits;
}

__global__ __launch_bounds__(256) void quantize_fp8_kernel(const float* __restrict__ in,
                                                           unsigned int* __restrict__ out,
                                                           const unsigned int* __restrict__ amaxBits,
                                                           size_t n4) {
    size_t i = (size_t)blockIdx.x * blockDim.x + threadIdx.x;
    if (i >= n4) return;
    float amax = fmaxf(__uint_as_float(*amaxBits), 1e-12f);
    float s    = FP8_MAX / amax;
    float4 v   = ((const float4*)in)[i];
    float a = fminf(fmaxf(v.x * s, -FP8_MAX), FP8_MAX);
    float b = fminf(fmaxf(v.y * s, -FP8_MAX), FP8_MAX);
    float c = fminf(fmaxf(v.z * s, -FP8_MAX), FP8_MAX);
    float d = fminf(fmaxf(v.w * s, -FP8_MAX), FP8_MAX);
    unsigned int packed = fp8_e4m3_bits(a)        |
                          (fp8_e4m3_bits(b) << 8) |
                          (fp8_e4m3_bits(c) << 16)|
                          (fp8_e4m3_bits(d) << 24);
    out[i] = packed;
}

__device__ __forceinline__ v16i load_a_frag(const unsigned char* __restrict__ rowPtr,
                                            int half) {
    v16i a;
#pragma unroll
    for (int j = 0; j < 4; ++j) {
        uint4 d = *(const uint4*)(rowPtr + 32 * j + 16 * half);
        a[4 * j + 0] = (int)d.x;
        a[4 * j + 1] = (int)d.y;
        a[4 * j + 2] = (int)d.z;
        a[4 * j + 3] = (int)d.w;
    }
    return a;
}

__device__ __forceinline__ v16i load_b_frag(const unsigned char* __restrict__ colPtr,
                                            int half) {
    v16i b;
#pragma unroll
    for (int c = 0; c < 4; ++c) {
        uint4 d = *(const uint4*)(colPtr + 32 * c + 16 * half);
        b[4 * c + 0] = (int)d.x;
        b[4 * c + 1] = (int)d.y;
        b[4 * c + 2] = (int)d.z;
        b[4 * c + 3] = (int)d.w;
    }
    return b;
}

__global__ __launch_bounds__(256) void fp8_gemm_wmma_kernel(
    const unsigned char* __restrict__ qx,
    const unsigned char* __restrict__ qw,
    const unsigned int*  __restrict__ amax,
    const float*         __restrict__ bias,
    float*               __restrict__ out) {

    const int lane  = threadIdx.x & 31;
    const int wave  = threadIdx.x >> 5;
    const int waveM = wave & 1;
    const int waveN = wave >> 1;
    const int lo    = lane & 15;
    const int half  = lane >> 4;

    const int mBase = blockIdx.y * 128 + waveM * 64;
    const int nBase = blockIdx.x * 256 + waveN * 64;

    const unsigned char* aRowBase = qx + (size_t)(mBase + lo) * Kdim;
    const unsigned char* bColBase = qw + (size_t)(nBase + lo) * Kdim;

    v8f acc[4][4] = {};

    for (int k = 0; k < Kdim; k += 128) {
        __builtin_prefetch(aRowBase + k + 128, 0, 1);
        __builtin_prefetch(bColBase + k + 128, 0, 1);

        v16i a[4];
#pragma unroll
        for (int tm = 0; tm < 4; ++tm)
            a[tm] = load_a_frag(aRowBase + (size_t)tm * 16 * Kdim + k, half);

#pragma unroll
        for (int tn = 0; tn < 4; ++tn) {
            v16i b = load_b_frag(bColBase + (size_t)tn * 16 * Kdim + k, half);
#pragma unroll
            for (int tm = 0; tm < 4; ++tm) {
                acc[tm][tn] = __builtin_amdgcn_wmma_f32_16x16x128_fp8_fp8(
                    a[tm], b, (short)0, acc[tm][tn], false, false);
            }
        }
    }

    const float ax = fmaxf(__uint_as_float(amax[0]), 1e-12f);
    const float aw = fmaxf(__uint_as_float(amax[1]), 1e-12f);
    const float sc = (ax * aw) * (1.0f / (FP8_MAX * FP8_MAX));

#pragma unroll
    for (int tm = 0; tm < 4; ++tm) {
#pragma unroll
        for (int tn = 0; tn < 4; ++tn) {
            const int col = nBase + tn * 16 + lo;
            const float bval = bias[col];
#pragma unroll
            for (int r = 0; r < 8; ++r) {
                const int row = mBase + tm * 16 + half * 8 + r;
                out[(size_t)row * Ndim + col] = acc[tm][tn][r] * sc + bval;
            }
        }
    }
}

extern "C" void kernel_launch(void* const* d_in, const int* in_sizes, int n_in,
                              void* d_out, int out_size, void* d_ws, size_t ws_size,
                              hipStream_t stream) {
    (void)in_sizes; (void)n_in; (void)out_size; (void)ws_size;

    const float* x    = (const float*)d_in[0];
    const float* w    = (const float*)d_in[1];
    const float* bias = (const float*)d_in[2];
    float*       out  = (float*)d_out;

    unsigned int*  amax = (unsigned int*)d_ws;
    unsigned char* qx   = (unsigned char*)d_ws + 256;
    unsigned char* qw   = qx + (size_t)Mdim * Kdim;

    const size_t nx4 = (size_t)Mdim * Kdim / 4;
    const size_t nw4 = (size_t)Ndim * Kdim / 4;

    init_amax_kernel<<<1, 32, 0, stream>>>(amax);

    amax_kernel<<<512, 256, 0, stream>>>(x, nx4, amax + 0);
    amax_kernel<<<1024, 256, 0, stream>>>(w, nw4, amax + 1);

    quantize_fp8_kernel<<<(int)((nx4 + 255) / 256), 256, 0, stream>>>(
        x, (unsigned int*)qx, amax + 0, nx4);
    quantize_fp8_kernel<<<(int)((nw4 + 255) / 256), 256, 0, stream>>>(
        w, (unsigned int*)qw, amax + 1, nw4);

    fp8_gemm_wmma_kernel<<<dim3(Ndim / 256, Mdim / 128), 256, 0, stream>>>(
        qx, qw, amax, bias, out);
}
